// unit_gcn_2net_72361609003088
// MI455X (gfx1250) — matrix operand order repaired and checked by reading, not compiled, not run on hardware
//
#include <hip/hip_runtime.h>

// ---------------------------------------------------------------------------
// Fused unit-GCN block for MI455X (gfx1250, wave32, WMMA bf16 16x16x32).
//
// One workgroup per (n, 8-timestep) tile; the whole 3-stage GCN block is
// (n,t)-local, so all intermediates live in LDS:
//   sx   : 256 ch x 256 cols (8t x 32 v-padded) bf16, row-major   = 131072 B
//   spk  : same 256x256 panel in packed WMMA-B fragment layout    = 131072 B
//   szp  : 64x256 subset scratch, packed WMMA-B layout            =  32768 B
//   sadj : 7 adjacency matrices, packed WMMA-B layout             =  14336 B
//   sbeta: fused bias vectors f32                                 =   1792 B
// Total ~304 KB -> needs CDNA5's 320KB-per-WGP LDS (1 WG/WGP).
//
// Packed B layout (K x N panel, 32x16 tiles): tile(kt,ct) = 1KB blob; lane l
// owns 32 contiguous bytes = u16[16], slot i = element (k = kt*32+16*(l>>4)+i,
// col = ct*16 + (l&15)).  B loads are 2 aligned ds_load_b128 (vs 16 u16
// gathers); producers scatter pairs with b32 stores (slot index == m&15).
//
// Prep kernel folds conv+BN algebraically: Wg_s=g3*(wg_s@wd_s),
// Wk_s=g5*(wk_s@wh_s), W2=g4*wf, WO=g2*w_out, fused biases, padded+packed
// adjacencies -> every stage is a pure bf16 GEMM with f32 WMMA accumulation.
// ---------------------------------------------------------------------------

typedef __attribute__((ext_vector_type(16))) __bf16 v16bf;
typedef __attribute__((ext_vector_type(8)))  float  v8f;

#define DEVINL __device__ __forceinline__

#define SXS 256       // LDS column stride (8 t * 32 padded v)
#define TILE_T 8
#define VV 25

// ws layout (ushort elements unless noted)
#define WS_WG   0         // 3*64*64
#define WS_WK   12288     // 3*64*64
#define WS_W2   24576     // 64*64
#define WS_WO   28672     // 256*256
#define WS_ADJ  94208     // 7*1024 packed (adj1[3], adj2, adj3[3])
#define WS_USHORTS 101376
#define WS_F32_BYTE_OFF (WS_USHORTS * 2)   // then 448 floats: beta1|beta2|beta3|betaO
#define PREP_TOTAL (WS_USHORTS + 448)

DEVINL unsigned short f2bf(float f) {
  union { float f; unsigned u; } x; x.f = f;
  unsigned r = x.u + 0x7FFFu + ((x.u >> 16) & 1u);   // RNE
  return (unsigned short)(r >> 16);
}
DEVINL float bf2f(unsigned short h) {
  union { unsigned u; float f; } x; x.u = ((unsigned)h) << 16;
  return x.f;
}
// two f32 -> packed bf16x2 dword (hardware v_cvt_pk_bf16_f32 when available)
DEVINL unsigned pack_bf16x2(float lo, float hi) {
#if defined(__gfx1250__) && __has_builtin(__builtin_amdgcn_cvt_pk_bf16_f32)
  auto t = __builtin_amdgcn_cvt_pk_bf16_f32(lo, hi);
  unsigned r; __builtin_memcpy(&r, &t, 4);
  return r;
#else
  return (unsigned)f2bf(lo) | ((unsigned)f2bf(hi) << 16);
#endif
}

DEVINL v8f wmma_bf16(v16bf a, v16bf b, v8f c) {
  return __builtin_amdgcn_wmma_f32_16x16x32_bf16(
      /*neg_a=*/false, a, /*neg_b=*/false, b,
      /*c_mod=*/(short)0, c, /*reuse_a=*/false, /*reuse_b=*/false);
}

// B fragment from packed layout: 2 aligned 16B LDS loads.
template <typename PT>
DEVINL v16bf load_B_packed(PT base, int nct, int kt, int ct, int lane) {
  union { v16bf v; uint4 q[2]; } f;
  auto p = base + (((kt * nct + ct) << 9) + (lane << 4));
  f.q[0] = *(const uint4*)(p);
  f.q[1] = *(const uint4*)(p + 8);
  return f.v;
}

// A fragment (16x32 bf16, M x K) from row-major storage (LDS or global):
// lane l -> row m = mbase+(l&15); VGPR0..3 = K 16h..16h+7, VGPR4..7 = K 16h+8..
template <typename PT>
DEVINL v16bf load_A_frag(PT base, int stride, int mbase, int kbase, int lane) {
  union { v16bf v; uint4 q[2]; } f;
  auto row = base + (mbase + (lane & 15)) * stride;
  const int h = (lane >> 4) << 4;
  f.q[0] = *(const uint4*)(row + kbase + h);
  f.q[1] = *(const uint4*)(row + kbase + h + 8);
  return f.v;
}

// Scatter a WMMA D fragment (f32) into a packed-B panel as bf16, paired b32
// stores (D rows m, m+1 land in adjacent u16 slots of the same lane chunk).
DEVINL void store_D_packed(unsigned short* base, int nct, int mBase, int colBase,
                           int lane, v8f d) {
  const int col  = colBase + (lane & 15);
  const int ct   = col >> 4;
  const int half = lane >> 4;
#pragma unroll
  for (int r = 0; r < 8; r += 2) {
    const int m  = mBase + r + (half << 3);
    const int kt = m >> 5;
    const int lo = (col & 15) | (((m >> 4) & 1) << 4);
    *(unsigned*)&base[(((kt * nct + ct) << 9) + (lo << 4) + (m & 15))] =
        pack_bf16x2(d[r], d[r + 1]);
  }
}

// dst rows += src rows (elementwise bf16, rows contiguous), 2 elems per dword
DEVINL void add_rows(unsigned short* sx, int dstRow, int srcRow, int tid) {
  unsigned* d = (unsigned*)&sx[dstRow * SXS];
  const unsigned* s = (const unsigned*)&sx[srcRow * SXS];
  for (int i = tid; i < 64 * SXS / 2; i += 256) {
    const unsigned dv = d[i], sv = s[i];
    d[i] = pack_bf16x2(bf2f((unsigned short)dv) + bf2f((unsigned short)sv),
                       bf2f((unsigned short)(dv >> 16)) + bf2f((unsigned short)(sv >> 16)));
  }
}

// One graph-conv stage: y = relu( sum_s W_s @ (X_chunk . Adj_s) + beta );
// result overwrites chunk rows of sx (row-major) AND the packed panel spk.
DEVINL void subset_stage(unsigned short* sx, unsigned short* szp, unsigned short* spk,
                         const unsigned short* adjPacked, int nsub,
                         const unsigned short* Wglb, const float* beta,
                         int chunkRow, int lane, int wave) {
  const v8f vzero = {0.f, 0.f, 0.f, 0.f, 0.f, 0.f, 0.f, 0.f};
  v8f acc[2][4];
#pragma unroll
  for (int ct = 0; ct < 2; ++ct)
#pragma unroll
    for (int rt = 0; rt < 4; ++rt) acc[ct][rt] = vzero;

  for (int s = 0; s < nsub; ++s) {
    // --- V-dim matmul: Z = X_chunk . Adj_s ; wave w owns timestep t = w ---
    v16bf adjB0 = load_B_packed(adjPacked + s * 1024, 2, 0, 0, lane);
    v16bf adjB1 = load_B_packed(adjPacked + s * 1024, 2, 0, 1, lane);
    const int t = wave;   // TILE_T == #waves == 8
#pragma unroll
    for (int rt = 0; rt < 4; ++rt) {
      v16bf a = load_A_frag(sx, SXS, chunkRow + rt * 16, t * 32, lane);
      v8f d0 = vzero, d1 = vzero;
      d0 = wmma_bf16(a, adjB0, d0);
      d1 = wmma_bf16(a, adjB1, d1);
      store_D_packed(szp, 16, rt * 16, t * 32,      lane, d0);
      store_D_packed(szp, 16, rt * 16, t * 32 + 16, lane, d1);
    }
    __syncthreads();

    // --- channel GEMM: acc += W_s(64x64) @ Z ; wave owns 2 column tiles ---
#pragma unroll
    for (int kt = 0; kt < 2; ++kt) {
      v16bf b0 = load_B_packed(szp, 16, kt, wave * 2 + 0, lane);
      v16bf b1 = load_B_packed(szp, 16, kt, wave * 2 + 1, lane);
#pragma unroll
      for (int rt = 0; rt < 4; ++rt) {
        v16bf a = load_A_frag(Wglb + s * 4096, 64, rt * 16, kt * 32, lane);
        acc[0][rt] = wmma_bf16(a, b0, acc[0][rt]);
        acc[1][rt] = wmma_bf16(a, b1, acc[1][rt]);
      }
    }
    __syncthreads();
  }

  // epilogue: relu(acc + beta) -> sx rows (row-major) + spk (packed)
#pragma unroll
  for (int ct = 0; ct < 2; ++ct) {
    const int col = (wave * 2 + ct) * 16 + (lane & 15);
#pragma unroll
    for (int rt = 0; rt < 4; ++rt) {
      const int mb = rt * 16 + ((lane >> 4) << 3);
#pragma unroll
      for (int r = 0; r < 8; r += 2) {
        const float v0 = fmaxf(acc[ct][rt][r]     + beta[mb + r],     0.f);
        const float v1 = fmaxf(acc[ct][rt][r + 1] + beta[mb + r + 1], 0.f);
        const unsigned p = pack_bf16x2(v0, v1);
        sx[(chunkRow + mb + r)     * SXS + col] = (unsigned short)p;
        sx[(chunkRow + mb + r + 1) * SXS + col] = (unsigned short)(p >> 16);
        const int m  = chunkRow + mb + r;               // even
        const int lo = (col & 15) | (((m >> 4) & 1) << 4);
        *(unsigned*)&spk[((((m >> 5) * 16 + (col >> 4)) << 9) + (lo << 4) + (m & 15))] = p;
      }
    }
  }
  __syncthreads();
}

// ---------------------------------------------------------------------------
// Prep kernel: fold conv/BN weights, pack adjacencies, build fused biases.
// ---------------------------------------------------------------------------
__global__ void gcn_prep(
    const float* __restrict__ A,   const float* __restrict__ PA1,
    const float* __restrict__ PA2, const float* __restrict__ PA3,
    const float* __restrict__ wd,  const float* __restrict__ bd,
    const float* __restrict__ wg,  const float* __restrict__ g3, const float* __restrict__ b3,
    const float* __restrict__ wf,  const float* __restrict__ g4, const float* __restrict__ b4,
    const float* __restrict__ wh,  const float* __restrict__ bh,
    const float* __restrict__ wk,  const float* __restrict__ g5, const float* __restrict__ b5,
    const float* __restrict__ w_out, const float* __restrict__ g2, const float* __restrict__ b2,
    unsigned short* __restrict__ wsh, float* __restrict__ wsf) {
  const int id = blockIdx.x * 256 + threadIdx.x;
  if (id < 12288) {                       // Wg_eff[s][o][c]
    const int s = id >> 12, o = (id >> 6) & 63, c = id & 63;
    float a = 0.f;
    for (int m = 0; m < 64; ++m) a += wg[o * 192 + s * 64 + m] * wd[s * 4096 + m * 64 + c];
    wsh[id] = f2bf(g3[o] * a);
  } else if (id < 24576) {                // Wk_eff[s][o][c]
    const int tt = id - 12288;
    const int s = tt >> 12, o = (tt >> 6) & 63, c = tt & 63;
    float a = 0.f;
    for (int m = 0; m < 64; ++m) a += wk[o * 192 + s * 64 + m] * wh[s * 4096 + m * 64 + c];
    wsh[id] = f2bf(g5[o] * a);
  } else if (id < 28672) {                // W2_eff
    const int tt = id - 24576, o = tt >> 6;
    wsh[id] = f2bf(g4[o] * wf[tt]);
  } else if (id < 94208) {                // WO_eff (256x256)
    const int tt = id - 28672, o = tt >> 8;
    wsh[id] = f2bf(g2[o] * w_out[tt]);
  } else if (id < WS_USHORTS) {           // adjacencies, packed WMMA-B layout
    const int tt = id - WS_ADJ;           // 0..7167 ; 1024 per matrix
    const int mat = tt >> 10;             // 0..2: A+PA1 ; 3: PA2 ; 4..6: A+PA3
    const int e = tt & 1023;
    const int lo = (e >> 4) & 31, slot = e & 15;
    const int w = ((e >> 9) << 4) | (lo & 15);     // column
    const int v = ((lo >> 4) << 4) | slot;         // k (row)
    float val = 0.f;
    if (v < VV && w < VV) {
      if (mat < 3)       val = A[mat * 625 + v * 25 + w] + PA1[mat * 625 + v * 25 + w];
      else if (mat == 3) val = PA2[v * 25 + w];
      else {
        const int s = mat - 4;
        val = A[s * 625 + v * 25 + w] + PA3[s * 625 + v * 25 + w];
      }
    }
    wsh[id] = f2bf(val);
  } else if (id < PREP_TOTAL) {           // fused bias vectors
    const int tt = id - WS_USHORTS;
    if (tt < 64) {                        // beta1 = b3 + g3*(wg @ bd)
      float a = 0.f;
      for (int j = 0; j < 192; ++j) a += wg[tt * 192 + j] * bd[j];
      wsf[tt] = b3[tt] + g3[tt] * a;
    } else if (tt < 128) {                // beta2 = b4
      wsf[tt] = b4[tt - 64];
    } else if (tt < 192) {                // beta3 = b5 + g5*(wk @ bh)
      const int o = tt - 128;
      float a = 0.f;
      for (int j = 0; j < 192; ++j) a += wk[o * 192 + j] * bh[j];
      wsf[tt] = b5[o] + g5[o] * a;
    } else {                              // betaO = b2
      wsf[tt] = b2[tt - 192];
    }
  }
}

// ---------------------------------------------------------------------------
// Main fused kernel: one workgroup per (n, 8-timestep) tile.
// ---------------------------------------------------------------------------
__global__ __launch_bounds__(256, 1) void gcn_main(
    const float* __restrict__ x, const unsigned short* __restrict__ wsh,
    const float* __restrict__ wsf, float* __restrict__ out) {
  __shared__ __align__(16) unsigned short sx[256 * SXS];    // row-major activations
  __shared__ __align__(16) unsigned short spk[8 * 16 * 512]; // packed final-GEMM panel
  __shared__ __align__(16) unsigned short szp[2 * 16 * 512]; // packed subset scratch
  __shared__ __align__(16) unsigned short sadj[7 * 1024];    // packed adjacencies
  __shared__ float sbeta[448];                               // beta1|beta2|beta3|betaO

  const int tid  = threadIdx.x;
  const int lane = tid & 31;
  const int wave = tid >> 5;
  const int n    = blockIdx.x >> 5;          // 32 time-groups per sample
  const int t0   = (blockIdx.x & 31) * TILE_T;

  const unsigned short* Wg = wsh + WS_WG;
  const unsigned short* Wk = wsh + WS_WK;
  const unsigned short* W2 = wsh + WS_W2;
  const unsigned short* WO = wsh + WS_WO;

  // keep the 128KB folded output weights hot in cache (global_prefetch_b8)
  for (int i = tid; i < (256 * 256 * 2) / 64; i += 256)
    __builtin_prefetch((const char*)WO + i * 64, 0, 1);

  for (int i = tid; i < 7 * 1024; i += 256) sadj[i] = wsh[WS_ADJ + i];
  for (int i = tid; i < 448;      i += 256) sbeta[i] = wsf[i];

  // load x tile -> bf16 LDS (channel, t*32+v) with v zero-padded;
  // xs0 rows (<64) additionally written to the packed final-GEMM panel.
  const int xbase = ((n * 256) * 256 + t0) * 25;
  for (int i = tid; i < 256 * 256; i += 256) {
    const int ch = i >> 8, c = i & 255, t = c >> 5, v = c & 31;
    float val = 0.f;
    if (v < VV) val = x[xbase + (ch * 256 + t) * 25 + v];
    const unsigned short h = f2bf(val);
    sx[ch * SXS + c] = h;
    if (ch < 64) {
      const int lo = (c & 15) | (((ch >> 4) & 1) << 4);
      spk[(((ch >> 5) * 16 + (c >> 4)) << 9) + (lo << 4) + (ch & 15)] = h;
    }
  }
  __syncthreads();

  // stage 1: y1 = relu(sum_s Wg_s @ (xs1 . (A+PA1)_s) + beta1)  -> rows 64..127
  subset_stage(sx, szp, spk, sadj,        3, Wg, sbeta,       64,  lane, wave);
  add_rows(sx, 128, 64, tid);                          // t2 = xs2 + y1
  __syncthreads();
  // stage 2: y2 = relu(W2 @ (t2 . PA2) + beta2)                 -> rows 128..191
  subset_stage(sx, szp, spk, sadj + 3072, 1, W2, sbeta + 64,  128, lane, wave);
  add_rows(sx, 192, 128, tid);                         // t3 = xs3 + y2
  __syncthreads();
  // stage 3: y3 = relu(sum_s Wk_s @ (t3 . (A+PA3)_s) + beta3)   -> rows 192..255
  subset_stage(sx, szp, spk, sadj + 4096, 3, Wk, sbeta + 128, 192, lane, wave);

  // final: out = relu( WO @ [xs0; y1; y2; y3] + betaO + x ), K = 256
  const v8f vzero = {0.f, 0.f, 0.f, 0.f, 0.f, 0.f, 0.f, 0.f};
#pragma unroll
  for (int ct = 0; ct < 2; ++ct) {
    const int ctile = wave * 2 + ct;
    v8f acc[4][4];
#pragma unroll
    for (int rg = 0; rg < 4; ++rg)
#pragma unroll
      for (int rt = 0; rt < 4; ++rt) acc[rg][rt] = vzero;

#pragma unroll
    for (int kt = 0; kt < 8; ++kt) {
      v16bf b = load_B_packed(spk, 16, kt, ctile, lane);   // feeds 16 WMMAs
#pragma unroll
      for (int rg = 0; rg < 4; ++rg)
#pragma unroll
        for (int rt = 0; rt < 4; ++rt) {
          v16bf a = load_A_frag(WO, 256, rg * 64 + rt * 16, kt * 32, lane);
          acc[rg][rt] = wmma_bf16(a, b, acc[rg][rt]);
        }
    }

    const int col = ctile * 16 + (lane & 15);
    const int t = col >> 5, v = col & 31;
    if (v < VV) {                                          // drop v-pad columns
      const int obase = n * (256 * 256 * 25) + (t0 + t) * 25 + v;
#pragma unroll
      for (int rg = 0; rg < 4; ++rg)
#pragma unroll
        for (int rt = 0; rt < 4; ++rt) {
          const int mb = rg * 64 + rt * 16 + ((lane >> 4) << 3);
#pragma unroll
          for (int r = 0; r < 8; ++r) {
            const int m = mb + r;
            const int idx = obase + m * (256 * 25);
            out[idx] = fmaxf(acc[rg][rt][r] + sbeta[192 + m] + x[idx], 0.f);
          }
        }
    }
  }
}

extern "C" void kernel_launch(void* const* d_in, const int* in_sizes, int n_in,
                              void* d_out, int out_size, void* d_ws, size_t ws_size,
                              hipStream_t stream) {
  const float* x     = (const float*)d_in[0];
  const float* A     = (const float*)d_in[1];
  const float* PA1   = (const float*)d_in[2];
  const float* PA2   = (const float*)d_in[3];
  const float* PA3   = (const float*)d_in[4];
  const float* wd    = (const float*)d_in[5];
  const float* bd    = (const float*)d_in[6];
  const float* wg    = (const float*)d_in[7];
  const float* g3    = (const float*)d_in[8];
  const float* b3    = (const float*)d_in[9];
  const float* wf    = (const float*)d_in[10];
  const float* g4    = (const float*)d_in[11];
  const float* b4    = (const float*)d_in[12];
  const float* wh    = (const float*)d_in[13];
  const float* bh    = (const float*)d_in[14];
  const float* wk    = (const float*)d_in[15];
  const float* g5    = (const float*)d_in[16];
  const float* b5    = (const float*)d_in[17];
  const float* w_out = (const float*)d_in[18];
  const float* g2    = (const float*)d_in[19];
  const float* b2    = (const float*)d_in[20];

  unsigned short* wsh = (unsigned short*)d_ws;
  float*          wsf = (float*)((char*)d_ws + WS_F32_BYTE_OFF);

  gcn_prep<<<(PREP_TOTAL + 255) / 256, 256, 0, stream>>>(
      A, PA1, PA2, PA3, wd, bd, wg, g3, b3, wf, g4, b4,
      wh, bh, wk, g5, b5, w_out, g2, b2, wsh, wsf);

  gcn_main<<<dim3(32 * (256 / TILE_T)), dim3(256), 0, stream>>>(
      x, wsh, wsf, (float*)d_out);
}
